// Spatial_Window_Attention_68427418960335
// MI455X (gfx1250) — matrix operand order repaired and checked by reading, not compiled, not run on hardware
//
#include <hip/hip_runtime.h>
#include <math.h>

// ---------------- constants ----------------
#define BATCH 4
#define CH    192
#define HH_   256
#define WW_   256
#define WS_   8
#define SHIFT_ 4
#define NHEAD 6
#define HD_   32
#define NTOK  64           // tokens per window
#define NWIN  1024         // windows per image
#define C3    576          // 3*C
#define HWPIX 65536        // H*W
#define MTOT  262144       // B*H*W  == B_*NTOK

typedef float v8f  __attribute__((ext_vector_type(8)));
typedef __bf16 v16bf __attribute__((ext_vector_type(16)));
typedef __bf16 bf8v  __attribute__((ext_vector_type(8)));
typedef unsigned int u32x4 __attribute__((ext_vector_type(4)));
typedef int i32x4 __attribute__((ext_vector_type(4)));
typedef int i32x8 __attribute__((ext_vector_type(8)));

// ---- Tensor Data Mover: 1-D copy of nElem bf16 elements global -> LDS ----
// D# per CDNA5 ISA: group0 {count=1, lds_addr, 57b
// global_addr, type=2}; group1 {data_size=2B, tensor_dim0=nElem, dim1=1,
// tile_dim0=nElem, tile_dim1=1, stride0=nElem}; groups 2/3 zero (2-D tensor).
__device__ __forceinline__ void tdm_load_1d(unsigned ldsOff, const void* gsrc,
                                            unsigned nElem /* <= 65535 */) {
  unsigned long long ga = (unsigned long long)gsrc;
  u32x4 g0;
  g0.x = 1u;                                           // count=1, user descriptor
  g0.y = ldsOff;                                       // LDS byte address
  g0.z = (unsigned)ga;                                 // global_addr[31:0]
  g0.w = ((unsigned)(ga >> 32) & 0x01ffffffu) | (2u << 30);  // addr[56:32] | type=2
  i32x8 g1;
  g1[0] = (int)(1u << 16);                             // data_size=1 (2 bytes)
  g1[1] = (int)((nElem & 0xffffu) << 16);              // tensor_dim0[15:0]
  g1[2] = (int)((nElem >> 16) | (1u << 16));           // tensor_dim0[31:16] | dim1=1
  g1[3] = (int)(nElem << 16);                          // dim1 hi=0 | tile_dim0
  g1[4] = 1;                                           // tile_dim1=1, tile_dim2=0
  g1[5] = (int)nElem;                                  // tensor_dim0_stride lo
  g1[6] = 0;
  g1[7] = 0;
  i32x4 z4 = {0, 0, 0, 0};
#if __clang_major__ >= 23
  i32x8 z8 = {0, 0, 0, 0, 0, 0, 0, 0};
  __builtin_amdgcn_tensor_load_to_lds(g0, g1, z4, z4, z8, 0);
#else
  __builtin_amdgcn_tensor_load_to_lds(g0, g1, z4, z4, 0);
#endif
}
__device__ __forceinline__ unsigned lds_off(const void* p) {
  return (unsigned)(unsigned long long)p;   // generic LDS ptr: low 32 bits = offset
}

// ---- WMMA fragment loaders (CDNA5 16x16x32 bf16 layouts) ----
__device__ __forceinline__ v16bf load_frag_A(const __bf16* rowPtr, int half) {
  bf8v c0 = *(const bf8v*)(rowPtr + half * 8);
  bf8v c1 = *(const bf8v*)(rowPtr + 16 + half * 8);
  v16bf r;
#pragma unroll
  for (int i = 0; i < 8; ++i) { r[i] = c0[i]; r[8 + i] = c1[i]; }
  return r;
}
__device__ __forceinline__ v16bf load_frag_B(const __bf16* colPtr, int half) {
  bf8v c0 = *(const bf8v*)(colPtr + half * 8);
  bf8v c1 = *(const bf8v*)(colPtr + 16 + half * 8);
  v16bf r;
#pragma unroll
  for (int i = 0; i < 8; ++i) { r[i] = c0[i]; r[8 + i] = c1[i]; }
  return r;
}

// ---------------- prep kernels ----------------
__global__ void cvt_f32_bf16(const float* __restrict__ s, __bf16* __restrict__ d, int n) {
  int i = blockIdx.x * 256 + threadIdx.x;
  if (i < n) d[i] = (__bf16)s[i];
}

__global__ void build_rpb(const float* __restrict__ tbl, float* __restrict__ rpb) {
  int i = blockIdx.x * 256 + threadIdx.x;           // NHEAD*64*64
  int m = i & 63, n = (i >> 6) & 63, head = i >> 12;
  int dh = (n >> 3) - (m >> 3) + 7;
  int dw = (n & 7) - (m & 7) + 7;
  rpb[i] = tbl[(dh * 15 + dw) * NHEAD + head];
}

__global__ void transpose_cvt(const float* __restrict__ in, __bf16* __restrict__ out) {
  size_t e = (size_t)blockIdx.x * 256 + threadIdx.x;
  int c = (int)(e % CH);
  size_t p = e / CH;
  int b = (int)(p / HWPIX);
  int hw = (int)(p % HWPIX);
  out[p * CH + c] = (__bf16)in[((size_t)(b * CH + c)) * HWPIX + hw];
}

// ---------------- conv1x1 as WMMA GEMM, weight panel staged via TDM ----------------
__global__ void conv1x1_wmma(const __bf16* __restrict__ inT,
                             const __bf16* __restrict__ wB,     // (576,192) bf16
                             const float*  __restrict__ bias,
                             __bf16* __restrict__ outT) {       // (M, 576) bf16
  __shared__ __align__(16) __bf16 sW[64 * CH];                  // 24 KB weight panel
  const int lane = threadIdx.x & 31;
  const int wave = threadIdx.x >> 5;
  const int half = lane >> 4;
  const int ml   = lane & 15;
  const int m0 = (blockIdx.x * 8 + wave) * 16;
  const int n0 = blockIdx.y * 64;

  if (threadIdx.x < 32) {                       // wave 0 issues the DMA
    tdm_load_1d(lds_off(sW), wB + (size_t)n0 * CH, 64 * CH);
    __builtin_amdgcn_s_wait_tensorcnt(0);
  }
  __syncthreads();

  v8f acc[4] = {};
#pragma unroll
  for (int k0 = 0; k0 < CH; k0 += 32) {
    v16bf a = load_frag_A(inT + (size_t)(m0 + ml) * CH + k0, half);
#pragma unroll
    for (int nt = 0; nt < 4; ++nt) {
      v16bf b = load_frag_B(sW + (size_t)(nt * 16 + ml) * CH + k0, half);
      acc[nt] = __builtin_amdgcn_wmma_f32_16x16x32_bf16(false, a, false, b,
                                                        (short)0, acc[nt], false, false);
    }
  }
#pragma unroll
  for (int nt = 0; nt < 4; ++nt) {
    int n = n0 + nt * 16 + ml;
    float bs = bias[n];
#pragma unroll
    for (int r = 0; r < 8; ++r) {
      int m = m0 + r + half * 8;
      outT[(size_t)m * C3 + n] = (__bf16)(acc[nt][r] + bs);
    }
  }
}

// ---------------- depthwise 3x3 + fuse + shift + partition ----------------
__global__ void dwconv_fuse(const __bf16* __restrict__ q1x, const __bf16* __restrict__ q1e,
                            const float* __restrict__ dwW,  const float* __restrict__ dwB,
                            const float* __restrict__ dweW, const float* __restrict__ dweB,
                            const float* __restrict__ a1p,  const float* __restrict__ a2p,
                            __bf16* __restrict__ Qb, __bf16* __restrict__ Kb,
                            __bf16* __restrict__ VTb) {
  long long idx = (long long)blockIdx.x * 256 + threadIdx.x;   // B*H*W*72
  int cg = (int)(idx % 72);
  long long p = idx / 72;
  int ww = (int)(p % WW_);
  long long t = p / WW_;
  int hh = (int)(t % HH_);
  int b  = (int)(t / HH_);
  int c0 = cg * 8;

  float accX[8] = {0.f,0.f,0.f,0.f,0.f,0.f,0.f,0.f};
  float accE[8] = {0.f,0.f,0.f,0.f,0.f,0.f,0.f,0.f};
#pragma unroll
  for (int dy = 0; dy < 3; ++dy) {
    int y = hh + dy - 1;
    if ((unsigned)y >= (unsigned)HH_) continue;
#pragma unroll
    for (int dx = 0; dx < 3; ++dx) {
      int xx = ww + dx - 1;
      if ((unsigned)xx >= (unsigned)WW_) continue;
      size_t base = ((size_t)((b * HH_ + y) * WW_ + xx)) * C3 + c0;
      bf8v vx = *(const bf8v*)(q1x + base);
      bf8v ve = *(const bf8v*)(q1e + base);
      int tap = dy * 3 + dx;
#pragma unroll
      for (int i = 0; i < 8; ++i) {
        accX[i] += (float)vx[i] * dwW[(c0 + i) * 9 + tap];
        accE[i] += (float)ve[i] * dweW[(c0 + i) * 9 + tap];
      }
    }
  }
  float a1 = *a1p, a2 = *a2p;
  int hs  = (hh + HH_ - SHIFT_) & (HH_ - 1);
  int ws2 = (ww + WW_ - SHIFT_) & (WW_ - 1);
  int b_  = b * NWIN + (hs >> 3) * 32 + (ws2 >> 3);
  int n   = (hs & 7) * 8 + (ws2 & 7);
  int which = c0 / CH;
  int cc = c0 % CH;
  int head = cc >> 5;
  int hd0  = cc & 31;

  if (which == 0) {
    const float scale = 0.17677669529663687f;   // HD^-0.5
    bf8v o;
#pragma unroll
    for (int i = 0; i < 8; ++i) o[i] = (__bf16)((accX[i] + dwB[c0 + i]) * scale);
    *(bf8v*)(Qb + (((size_t)b_ * NHEAD + head) * NTOK + n) * HD_ + hd0) = o;
  } else if (which == 1) {
    bf8v o;
#pragma unroll
    for (int i = 0; i < 8; ++i)
      o[i] = (__bf16)((accX[i] + dwB[c0 + i]) + a1 * (accE[i] + dweB[c0 + i]));
    *(bf8v*)(Kb + (((size_t)b_ * NHEAD + head) * NTOK + n) * HD_ + hd0) = o;
  } else {
#pragma unroll
    for (int i = 0; i < 8; ++i)
      VTb[(((size_t)b_ * NHEAD + head) * HD_ + hd0 + i) * NTOK + n] =
          (__bf16)((accX[i] + dwB[c0 + i]) + a2 * (accE[i] + dweB[c0 + i]));
  }
}

// ---------------- attention: one (window, head) per 2-wave block ----------------
#define SS_LD 65   // f32 score row stride (bank-spread)
#define SP_LD 72   // bf16 prob row stride (16B-aligned chunks)
__global__ void attn_wmma(const __bf16* __restrict__ Qb, const __bf16* __restrict__ Kb,
                          const __bf16* __restrict__ VTb,
                          const float* __restrict__ rpb, const float* __restrict__ mask,
                          __bf16* __restrict__ outA) {
  __shared__ float sS[NTOK * SS_LD];
  __shared__ __align__(16) __bf16 sP[NTOK * SP_LD];
  __shared__ __align__(16) __bf16 sQ[NTOK * HD_];
  __shared__ __align__(16) __bf16 sK[NTOK * HD_];
  __shared__ __align__(16) __bf16 sV[HD_ * NTOK];

  const int b_   = blockIdx.x / NHEAD;
  const int head = blockIdx.x % NHEAD;
  const int wi   = b_ & (NWIN - 1);
  const int lane = threadIdx.x & 31;
  const int wave = threadIdx.x >> 5;     // 0..1
  const int half = lane >> 4;
  const int ml   = lane & 15;

  const __bf16* Qp = Qb  + ((size_t)b_ * NHEAD + head) * NTOK * HD_;
  const __bf16* Kp = Kb  + ((size_t)b_ * NHEAD + head) * NTOK * HD_;
  const __bf16* Vp = VTb + ((size_t)b_ * NHEAD + head) * HD_ * NTOK;

  // ---- stage Q, K, V^T tiles (4 KB each) into LDS via the TDM ----
  if (threadIdx.x < 32) {
    tdm_load_1d(lds_off(sQ), Qp, NTOK * HD_);
    tdm_load_1d(lds_off(sK), Kp, NTOK * HD_);
    tdm_load_1d(lds_off(sV), Vp, HD_ * NTOK);
    __builtin_amdgcn_s_wait_tensorcnt(0);
  }
  __syncthreads();

  // ---- S = Q @ K^T ----
  v16bf aq[2];
#pragma unroll
  for (int mt = 0; mt < 2; ++mt)
    aq[mt] = load_frag_A(sQ + (wave * 32 + mt * 16 + ml) * HD_, half);
  v8f acc[2][4] = {};
#pragma unroll
  for (int nt = 0; nt < 4; ++nt) {
    v16bf bk = load_frag_B(sK + (nt * 16 + ml) * HD_, half);
#pragma unroll
    for (int mt = 0; mt < 2; ++mt)
      acc[mt][nt] = __builtin_amdgcn_wmma_f32_16x16x32_bf16(false, aq[mt], false, bk,
                                                            (short)0, acc[mt][nt], false, false);
  }
  const float* rpbH  = rpb  + (size_t)head * NTOK * NTOK;
  const float* maskW = mask + (size_t)wi * NTOK * NTOK;
#pragma unroll
  for (int mt = 0; mt < 2; ++mt)
#pragma unroll
    for (int nt = 0; nt < 4; ++nt)
#pragma unroll
      for (int r = 0; r < 8; ++r) {
        int row = wave * 32 + mt * 16 + r + half * 8;
        int col = nt * 16 + ml;
        sS[row * SS_LD + col] = acc[mt][nt][r] + rpbH[row * 64 + col] + maskW[row * 64 + col];
      }
  __syncthreads();

  // ---- softmax: one row per thread ----
  {
    int row = threadIdx.x;
    float mx = -1e30f;
#pragma unroll 8
    for (int j = 0; j < 64; ++j) mx = fmaxf(mx, sS[row * SS_LD + j]);
    float sum = 0.f;
#pragma unroll 8
    for (int j = 0; j < 64; ++j) {
      float e = __expf(sS[row * SS_LD + j] - mx);
      sS[row * SS_LD + j] = e;
      sum += e;
    }
    float inv = 1.f / sum;
#pragma unroll 8
    for (int j = 0; j < 64; ++j)
      sP[row * SP_LD + j] = (__bf16)(sS[row * SS_LD + j] * inv);
  }
  __syncthreads();

  // ---- O = P @ V ----
  v8f oacc[2][2] = {};
#pragma unroll
  for (int kt = 0; kt < 2; ++kt) {
    v16bf aP[2];
#pragma unroll
    for (int mt = 0; mt < 2; ++mt)
      aP[mt] = load_frag_A(&sP[(wave * 32 + mt * 16 + ml) * SP_LD + kt * 32], half);
#pragma unroll
    for (int nt = 0; nt < 2; ++nt) {
      v16bf bv = load_frag_B(sV + (nt * 16 + ml) * NTOK + kt * 32, half);
#pragma unroll
      for (int mt = 0; mt < 2; ++mt)
        oacc[mt][nt] = __builtin_amdgcn_wmma_f32_16x16x32_bf16(false, aP[mt], false, bv,
                                                               (short)0, oacc[mt][nt], false, false);
    }
  }
#pragma unroll
  for (int mt = 0; mt < 2; ++mt)
#pragma unroll
    for (int nt = 0; nt < 2; ++nt)
#pragma unroll
      for (int r = 0; r < 8; ++r) {
        int tok = wave * 32 + mt * 16 + r + half * 8;
        int hd  = nt * 16 + ml;
        outA[((size_t)b_ * NTOK + tok) * CH + head * HD_ + hd] = (__bf16)oacc[mt][nt][r];
      }
}

// ---------------- proj GEMM + window-reverse + un-shift + NCHW store ----------------
__global__ void proj_wmma(const __bf16* __restrict__ A,      // (M,192) bf16
                          const __bf16* __restrict__ Wp,     // (192,192) bf16 (o-major)
                          const float* __restrict__ pb,
                          float* __restrict__ out) {         // (B,C,H,W) f32
  __shared__ __align__(16) __bf16 sW[64 * CH];
  const int lane = threadIdx.x & 31;
  const int wave = threadIdx.x >> 5;
  const int half = lane >> 4;
  const int ml   = lane & 15;
  const int m0 = (blockIdx.x * 8 + wave) * 16;
  const int n0 = blockIdx.y * 64;

  if (threadIdx.x < 32) {
    tdm_load_1d(lds_off(sW), Wp + (size_t)n0 * CH, 64 * CH);
    __builtin_amdgcn_s_wait_tensorcnt(0);
  }
  __syncthreads();

  v8f acc[4] = {};
#pragma unroll
  for (int k0 = 0; k0 < CH; k0 += 32) {
    v16bf a = load_frag_A(A + (size_t)(m0 + ml) * CH + k0, half);
#pragma unroll
    for (int nt = 0; nt < 4; ++nt) {
      v16bf b = load_frag_B(sW + (size_t)(nt * 16 + ml) * CH + k0, half);
      acc[nt] = __builtin_amdgcn_wmma_f32_16x16x32_bf16(false, a, false, b,
                                                        (short)0, acc[nt], false, false);
    }
  }
#pragma unroll
  for (int nt = 0; nt < 4; ++nt) {
    int o = n0 + nt * 16 + ml;
    float bs = pb[o];
#pragma unroll
    for (int r = 0; r < 8; ++r) {
      int m  = m0 + r + half * 8;
      int b_ = m >> 6, n = m & 63;
      int b  = b_ >> 10, wiD = b_ & 1023;
      int hs = ((wiD >> 5) << 3) + (n >> 3);
      int ws2 = ((wiD & 31) << 3) + (n & 7);
      int h = (hs + SHIFT_) & (HH_ - 1);
      int w = (ws2 + SHIFT_) & (WW_ - 1);
      out[(((size_t)b * CH + o) << 16) + (h << 8) + w] = acc[nt][r] + bs;
    }
  }
}

// ---------------- host launcher ----------------
extern "C" void kernel_launch(void* const* d_in, const int* in_sizes, int n_in,
                              void* d_out, int out_size, void* d_ws, size_t ws_size,
                              hipStream_t stream) {
  const float* x       = (const float*)d_in[0];
  const float* edge    = (const float*)d_in[1];
  const float* mask    = (const float*)d_in[2];
  const float* qkv_w   = (const float*)d_in[3];
  const float* qkv_b   = (const float*)d_in[4];
  const float* dw_w    = (const float*)d_in[5];
  const float* dw_b    = (const float*)d_in[6];
  const float* qkv_e_w = (const float*)d_in[7];
  const float* qkv_e_b = (const float*)d_in[8];
  const float* dw_e_w  = (const float*)d_in[9];
  const float* dw_e_b  = (const float*)d_in[10];
  const float* rpb_tab = (const float*)d_in[11];
  const float* alpha1  = (const float*)d_in[12];
  const float* alpha2  = (const float*)d_in[13];
  const float* proj_w  = (const float*)d_in[14];
  const float* proj_b  = (const float*)d_in[15];

  char* ws = (char*)d_ws;
  const size_t SZ16 = (size_t)MTOT * CH * 2;   // 96 MB
  const size_t SZ48 = (size_t)MTOT * C3 * 2;   // 288 MB
  __bf16* xT   = (__bf16*)(ws + 0);
  __bf16* eT   = (__bf16*)(ws + SZ16);
  __bf16* q1x  = (__bf16*)(ws + 2 * SZ16);
  __bf16* q1e  = (__bf16*)(ws + 2 * SZ16 + SZ48);
  __bf16* VTb  = (__bf16*)(ws + 2 * SZ16 + 2 * SZ48);
  char*   wsm  = ws + 2 * SZ16 + 2 * SZ48 + SZ16;
  __bf16* wq   = (__bf16*)(wsm);                         // 576*192
  __bf16* we   = (__bf16*)(wsm + 221184);
  __bf16* wp   = (__bf16*)(wsm + 2 * 221184);            // 192*192
  float*  rpb  = (float*)(wsm + 2 * 221184 + 73728 + 256 - ((2*221184+73728) & 255));
  __bf16* Qb      = xT;    // region reuse after conv1x1 consumed the transposes
  __bf16* Kb      = eT;
  __bf16* attnOut = q1x;

  cvt_f32_bf16<<<(C3 * CH + 255) / 256, 256, 0, stream>>>(qkv_w,   wq, C3 * CH);
  cvt_f32_bf16<<<(C3 * CH + 255) / 256, 256, 0, stream>>>(qkv_e_w, we, C3 * CH);
  cvt_f32_bf16<<<(CH * CH + 255) / 256, 256, 0, stream>>>(proj_w,  wp, CH * CH);
  build_rpb<<<(NHEAD * NTOK * NTOK) / 256, 256, 0, stream>>>(rpb_tab, rpb);

  transpose_cvt<<<(unsigned)(((size_t)MTOT * CH) / 256), 256, 0, stream>>>(x,    xT);
  transpose_cvt<<<(unsigned)(((size_t)MTOT * CH) / 256), 256, 0, stream>>>(edge, eT);

  dim3 g1(MTOT / (16 * 8), C3 / 64);
  conv1x1_wmma<<<g1, 256, 0, stream>>>(xT, wq, qkv_b,   q1x);
  conv1x1_wmma<<<g1, 256, 0, stream>>>(eT, we, qkv_e_b, q1e);

  dwconv_fuse<<<(unsigned)(((size_t)MTOT * 72) / 256), 256, 0, stream>>>(
      q1x, q1e, dw_w, dw_b, dw_e_w, dw_e_b, alpha1, alpha2, Qb, Kb, VTb);

  attn_wmma<<<BATCH * NWIN * NHEAD, 64, 0, stream>>>(Qb, Kb, VTb, rpb, mask, attnOut);

  dim3 g4(MTOT / (16 * 8), CH / 64);
  proj_wmma<<<g4, 256, 0, stream>>>(attnOut, wp, proj_b, (float*)d_out);
}
